// PC_Module_56444460204537
// MI455X (gfx1250) — matrix operand order repaired and checked by reading, not compiled, not run on hardware
//
#include <hip/hip_runtime.h>
#include <hip/hip_bf16.h>
#include <stdint.h>

#define B_  4
#define C_  64
#define N_  4096
#define CQ_ 8
#define EPS_ 1e-5f

typedef __attribute__((ext_vector_type(16))) __bf16 v16bf;
typedef __attribute__((ext_vector_type(8)))  float  v8f;
typedef __attribute__((ext_vector_type(2)))  float  v2f;

union Frag {
    v16bf v;
    uint32_t u[8];
    uint4 q[2];
    unsigned short s[16];
};

__device__ __forceinline__ unsigned short f2bf(float f) {
    union { float f; uint32_t u; } c; c.f = f;
    uint32_t r = c.u + 0x7FFFu + ((c.u >> 16) & 1u);   // round-to-nearest-even
    return (unsigned short)(r >> 16);
}

__device__ __forceinline__ v8f wmma_bf16(v16bf a, v16bf b, v8f c) {
    // D(16x16,f32) = A(16x32,bf16) x B(32x16,bf16) + C
    return __builtin_amdgcn_wmma_f32_16x16x32_bf16(false, a, false, b, (short)0, c, false, false);
}

__device__ __forceinline__ v8f wmma_f32k4(v2f a, v2f b, v8f c) {
    // D(16x16,f32) = A(16x4,f32) x B(4x16,f32) + C   -- full f32, K=4
    return __builtin_amdgcn_wmma_f32_16x16x4_f32(false, a, false, b, (short)0, c, false, false);
}

// ---------------------------------------------------------------------------
// Kernel 1: q/k/v projections.  One thread per (b, n).
//   qf, kf: f32  [B][N][8]  (row-major over d, 32B rows -> aligned v2f loads)
//   vb    : bf16 [B][C][N]
// ---------------------------------------------------------------------------
__global__ void proj_kernel(const float* __restrict__ x,
                            const float* __restrict__ Wq, const float* __restrict__ bq,
                            const float* __restrict__ Wk, const float* __restrict__ bk,
                            const float* __restrict__ Wv, const float* __restrict__ bv,
                            float* __restrict__ qf,
                            float* __restrict__ kf,
                            unsigned short* __restrict__ vb) {
    int gid = blockIdx.x * blockDim.x + threadIdx.x;   // 0 .. B*N-1
    int b = gid >> 12;
    int n = gid & (N_ - 1);
    float qa[CQ_] = {}, ka[CQ_] = {}, va[C_] = {};
    const float* xb = x + ((size_t)b * C_) * N_ + n;
    for (int c = 0; c < C_; ++c) {
        float xc = xb[(size_t)c * N_];
        #pragma unroll
        for (int d = 0; d < CQ_; ++d) {
            qa[d] = fmaf(Wq[d * C_ + c], xc, qa[d]);
            ka[d] = fmaf(Wk[d * C_ + c], xc, ka[d]);
        }
        #pragma unroll
        for (int d = 0; d < C_; ++d)
            va[d] = fmaf(Wv[d * C_ + c], xc, va[d]);
    }
    float* qrow = qf + (size_t)gid * CQ_;
    float* krow = kf + (size_t)gid * CQ_;
    #pragma unroll
    for (int d = 0; d < CQ_; ++d) {
        qrow[d] = qa[d] + bq[d];
        krow[d] = ka[d] + bk[d];
    }
    #pragma unroll
    for (int d = 0; d < C_; ++d)
        vb[((size_t)b * C_ + d) * N_ + n] = f2bf(va[d] + bv[d]);
}

// ---------------------------------------------------------------------------
// Kernel 2: fused PAM attention (flash style, two passes over n).
// One wave per (b, 16-row query tile).
//   energy via f32 16x16x4 WMMA (K=8 exact, all lanes live),
//   PV via bf16 16x16x32 WMMA with p transposed through LDS.
// outp[b,c,m] = sum_n v[c,n] * softmax_n(q[m]·k[n])
// ---------------------------------------------------------------------------
__global__ void __launch_bounds__(32)
pam_attn_kernel(const float* __restrict__ qf,
                const float* __restrict__ kfp,
                const unsigned short* __restrict__ vb,
                float* __restrict__ outp) {
    __shared__ unsigned short pl[16 * 32];   // p tile, [m][n_local]
    __shared__ float ls[16];

    int b    = blockIdx.x >> 8;
    int m0   = (blockIdx.x & 255) * 16;
    int lane = threadIdx.x;
    int half = lane >> 4;
    int lm   = lane & 15;

    // A fragments (f32 16x4): VGPR0/1 = K = {2*half, 2*half+1} (+kbase)
    const float* qp = qf + ((size_t)b * N_ + m0 + lm) * CQ_;
    v2f a0 = *(const v2f*)(qp + 2 * half);          // K 0..3
    v2f a1 = *(const v2f*)(qp + 4 + 2 * half);      // K 4..7

    const float* kp = kfp + (size_t)b * N_ * CQ_;
    const unsigned short* vbb = vb + (size_t)b * C_ * N_;

    v8f zero = {};
    float emax[8];
    #pragma unroll
    for (int r = 0; r < 8; ++r) emax[r] = -1e30f;

    // ---- pass 1: per-row maxima ----
    for (int nc = 0; nc < N_ / 32; ++nc) {
        int nb = nc * 32;
        const float* k0p = kp + (size_t)(nb + lm) * CQ_;
        const float* k1p = kp + (size_t)(nb + 16 + lm) * CQ_;
        v2f b00 = *(const v2f*)(k0p + 2 * half);
        v2f b01 = *(const v2f*)(k0p + 4 + 2 * half);
        v2f b10 = *(const v2f*)(k1p + 2 * half);
        v2f b11 = *(const v2f*)(k1p + 4 + 2 * half);
        v8f e0 = wmma_f32k4(a1, b01, wmma_f32k4(a0, b00, zero));
        v8f e1 = wmma_f32k4(a1, b11, wmma_f32k4(a0, b10, zero));
        #pragma unroll
        for (int r = 0; r < 8; ++r)
            emax[r] = fmaxf(emax[r], fmaxf(e0[r], e1[r]));
    }
    #pragma unroll
    for (int r = 0; r < 8; ++r) {
        #pragma unroll
        for (int mask = 1; mask < 16; mask <<= 1)
            emax[r] = fmaxf(emax[r], __shfl_xor(emax[r], mask));
    }

    // ---- pass 2: exp + PV accumulate ----
    float lsum[8];
    #pragma unroll
    for (int r = 0; r < 8; ++r) lsum[r] = 0.f;
    v8f acc[4];
    #pragma unroll
    for (int t = 0; t < 4; ++t) acc[t] = zero;

    for (int nc = 0; nc < N_ / 32; ++nc) {
        int nb = nc * 32;
        const float* k0p = kp + (size_t)(nb + lm) * CQ_;
        const float* k1p = kp + (size_t)(nb + 16 + lm) * CQ_;
        v2f b00 = *(const v2f*)(k0p + 2 * half);
        v2f b01 = *(const v2f*)(k0p + 4 + 2 * half);
        v2f b10 = *(const v2f*)(k1p + 2 * half);
        v2f b11 = *(const v2f*)(k1p + 4 + 2 * half);
        v8f e0 = wmma_f32k4(a1, b01, wmma_f32k4(a0, b00, zero));
        v8f e1 = wmma_f32k4(a1, b11, wmma_f32k4(a0, b10, zero));

        #pragma unroll
        for (int r = 0; r < 8; ++r) {
            float p0 = __expf(e0[r] - emax[r]);
            float p1 = __expf(e1[r] - emax[r]);
            lsum[r] += p0 + p1;
            int m = r + 8 * half;                 // D layout: row = vgpr + 8*(lane/16)
            pl[m * 32 + lm]      = f2bf(p0);
            pl[m * 32 + 16 + lm] = f2bf(p1);
        }
        __syncthreads();
        // B fragment of p^T: lane col = m (lm); K(n_local) = 16*half + 0..15
        Frag pf;
        pf.q[0] = *(const uint4*)(pl + lm * 32 + half * 16);
        pf.q[1] = *(const uint4*)(pl + lm * 32 + half * 16 + 8);
        __syncthreads();

        #pragma unroll
        for (int ct = 0; ct < 4; ++ct) {
            Frag vf;
            const unsigned short* vr = vbb + (size_t)(ct * 16 + lm) * N_ + nb;
            vf.q[0] = *(const uint4*)(vr + 16 * half);
            vf.q[1] = *(const uint4*)(vr + 16 * half + 8);
            acc[ct] = wmma_bf16(vf.v, pf.v, acc[ct]);
        }
    }

    // row sums -> per-lane-m layout via LDS
    #pragma unroll
    for (int r = 0; r < 8; ++r) {
        #pragma unroll
        for (int mask = 1; mask < 16; mask <<= 1)
            lsum[r] += __shfl_xor(lsum[r], mask);
    }
    #pragma unroll
    for (int r = 0; r < 8; ++r) ls[r + 8 * half] = lsum[r];
    __syncthreads();
    float invl = 1.0f / ls[lm];

    float* ob = outp + (size_t)b * C_ * N_;
    #pragma unroll
    for (int ct = 0; ct < 4; ++ct)
        #pragma unroll
        for (int r = 0; r < 8; ++r)
            ob[(size_t)(ct * 16 + r + 8 * half) * N_ + m0 + lm] = acc[ct][r] * invl;
}

// ---------------------------------------------------------------------------
// Kernel 3: y = BN(g * src) + res  (training-mode BN, biased var, per-channel
// stats over B*N).  One block per channel.  Also emits bf16 copy of y.
// ---------------------------------------------------------------------------
__global__ void bn_add_kernel(const float* __restrict__ src, const float* __restrict__ res,
                              const float* __restrict__ gscal, const float* __restrict__ w,
                              const float* __restrict__ bias,
                              float* __restrict__ yout, unsigned short* __restrict__ ybf) {
    __shared__ float s1[256], s2[256];
    int c = blockIdx.x, tid = threadIdx.x;
    float g = gscal[0];
    float sum = 0.f, sq = 0.f;
    for (int i = tid; i < B_ * N_; i += 256) {
        int b = i >> 12, n = i & (N_ - 1);
        float v = g * src[((size_t)(b * C_ + c)) * N_ + n];
        sum += v; sq += v * v;
    }
    s1[tid] = sum; s2[tid] = sq;
    __syncthreads();
    for (int off = 128; off > 0; off >>= 1) {
        if (tid < off) { s1[tid] += s1[tid + off]; s2[tid] += s2[tid + off]; }
        __syncthreads();
    }
    float mean  = s1[0] / (float)(B_ * N_);
    float var   = s2[0] / (float)(B_ * N_) - mean * mean;
    float scale = w[c] * rsqrtf(var + EPS_);
    float shift = bias[c] - mean * scale;
    for (int i = tid; i < B_ * N_; i += 256) {
        int b = i >> 12, n = i & (N_ - 1);
        size_t idx = ((size_t)(b * C_ + c)) * N_ + n;
        float v = fmaf(g * src[idx], scale, shift) + res[idx];
        yout[idx] = v;
        ybf[idx]  = f2bf(v);
    }
}

// ---------------------------------------------------------------------------
// Kernel 4: CAM energy  ec[b,i,j] = sum_n y[i,n]*y[j,n]   (K = 4096, WMMA)
// One wave per (b, 16x16 tile).
// ---------------------------------------------------------------------------
__global__ void __launch_bounds__(32)
cam_energy_kernel(const unsigned short* __restrict__ yb, float* __restrict__ ec) {
    int b  = blockIdx.x >> 4;
    int t  = blockIdx.x & 15;
    int it = t >> 2, jt = t & 3;
    int lane = threadIdx.x, half = lane >> 4, lm = lane & 15;
    const unsigned short* ybb = yb + (size_t)b * C_ * N_;
    v8f acc = {};
    for (int kc = 0; kc < N_ / 32; ++kc) {
        int k0 = kc * 32;
        Frag af, bf;
        const unsigned short* ar = ybb + (size_t)(it * 16 + lm) * N_ + k0;
        af.q[0] = *(const uint4*)(ar + 16 * half);
        af.q[1] = *(const uint4*)(ar + 16 * half + 8);
        const unsigned short* br = ybb + (size_t)(jt * 16 + lm) * N_ + k0 + 16 * half;
        bf.q[0] = *(const uint4*)(br);
        bf.q[1] = *(const uint4*)(br + 8);
        acc = wmma_bf16(af.v, bf.v, acc);
    }
    float* eb = ec + (size_t)b * C_ * C_;
    #pragma unroll
    for (int r = 0; r < 8; ++r)
        eb[(size_t)(it * 16 + r + 8 * half) * C_ + jt * 16 + lm] = acc[r];
}

// ---------------------------------------------------------------------------
// Kernel 5: CAM softmax over last dim of (rowmax - e).
// softmax(max-e) = exp(min - e) / sum  -> only the row min is needed.
// ---------------------------------------------------------------------------
__global__ void cam_softmax_kernel(const float* __restrict__ ec,
                                   unsigned short* __restrict__ attn) {
    int b = blockIdx.x, c = threadIdx.x;
    const float* row = ec + ((size_t)b * C_ + c) * C_;
    float mn = 1e30f;
    for (int d = 0; d < C_; ++d) mn = fminf(mn, row[d]);
    float s = 0.f;
    for (int d = 0; d < C_; ++d) s += __expf(mn - row[d]);
    float inv = 1.f / s;
    unsigned short* ar = attn + ((size_t)b * C_ + c) * C_;
    for (int d = 0; d < C_; ++d) ar[d] = f2bf(__expf(mn - row[d]) * inv);
}

// ---------------------------------------------------------------------------
// Kernel 6: out_c[b,c,n] = sum_d attn[c,d] * y[d,n]   (K = 64, WMMA)
// One wave per (b, 16-col n tile); covers all 4 c-tiles.
// ---------------------------------------------------------------------------
__global__ void __launch_bounds__(32)
cam_out_kernel(const unsigned short* __restrict__ attn,
               const unsigned short* __restrict__ yb,
               float* __restrict__ outc) {
    int b  = blockIdx.x >> 8;
    int n0 = (blockIdx.x & 255) * 16;
    int lane = threadIdx.x, half = lane >> 4, lm = lane & 15;
    const unsigned short* ybb = yb + (size_t)b * C_ * N_;
    const unsigned short* ab  = attn + (size_t)b * C_ * C_;
    v8f acc[4];
    #pragma unroll
    for (int t = 0; t < 4; ++t) acc[t] = v8f{};
    #pragma unroll
    for (int kc = 0; kc < 2; ++kc) {
        int k0 = kc * 32;
        Frag bf;   // B[k=d][col n] = y[d][n]; per-element strided, lane-coalesced
        #pragma unroll
        for (int e = 0; e < 16; ++e) {
            int d = k0 + 16 * half + e;
            bf.s[e] = ybb[(size_t)d * N_ + n0 + lm];
        }
        #pragma unroll
        for (int ct = 0; ct < 4; ++ct) {
            Frag af;
            const unsigned short* ar = ab + (size_t)(ct * 16 + lm) * C_ + k0;
            af.q[0] = *(const uint4*)(ar + 16 * half);
            af.q[1] = *(const uint4*)(ar + 16 * half + 8);
            acc[ct] = wmma_bf16(af.v, bf.v, acc[ct]);
        }
    }
    float* ob = outc + (size_t)b * C_ * N_;
    #pragma unroll
    for (int ct = 0; ct < 4; ++ct)
        #pragma unroll
        for (int r = 0; r < 8; ++r)
            ob[(size_t)(ct * 16 + r + 8 * half) * N_ + n0 + lm] = acc[ct][r];
}

// ---------------------------------------------------------------------------
extern "C" void kernel_launch(void* const* d_in, const int* in_sizes, int n_in,
                              void* d_out, int out_size, void* d_ws, size_t ws_size,
                              hipStream_t stream) {
    const float* x       = (const float*)d_in[0];
    const float* Wq      = (const float*)d_in[1];
    const float* bq      = (const float*)d_in[2];
    const float* Wk      = (const float*)d_in[3];
    const float* bk      = (const float*)d_in[4];
    const float* Wv      = (const float*)d_in[5];
    const float* bv      = (const float*)d_in[6];
    const float* gamma_p = (const float*)d_in[7];
    const float* bnp_w   = (const float*)d_in[8];
    const float* bnp_b   = (const float*)d_in[9];
    const float* gamma_c = (const float*)d_in[10];
    const float* bnc_w   = (const float*)d_in[11];
    const float* bnc_b   = (const float*)d_in[12];

    char* ws = (char*)d_ws;
    size_t off = 0;
    auto alloc = [&](size_t bytes) -> void* {
        void* p = ws + off;
        off += (bytes + 255) & ~(size_t)255;
        return p;
    };
    float*          qfl   = (float*)alloc((size_t)B_ * N_ * CQ_ * 4);
    float*          kfl   = (float*)alloc((size_t)B_ * N_ * CQ_ * 4);
    unsigned short* vb    = (unsigned short*)alloc((size_t)B_ * C_ * N_ * 2);
    float*          outp  = (float*)alloc((size_t)B_ * C_ * N_ * 4);
    float*          yf    = (float*)alloc((size_t)B_ * C_ * N_ * 4);
    unsigned short* ybf   = (unsigned short*)alloc((size_t)B_ * C_ * N_ * 2);
    float*          ec    = (float*)alloc((size_t)B_ * C_ * C_ * 4);
    unsigned short* attnc = (unsigned short*)alloc((size_t)B_ * C_ * C_ * 2);
    float*          outc  = (float*)alloc((size_t)B_ * C_ * N_ * 4);
    unsigned short* ybf2  = (unsigned short*)alloc((size_t)B_ * C_ * N_ * 2);

    // 1) projections
    proj_kernel<<<(B_ * N_) / 128, 128, 0, stream>>>(x, Wq, bq, Wk, bk, Wv, bv,
                                                     qfl, kfl, vb);
    // 2) fused PAM attention
    pam_attn_kernel<<<B_ * (N_ / 16), 32, 0, stream>>>(qfl, kfl, vb, outp);
    // 3) y = BN(gamma_p * outp) + x
    bn_add_kernel<<<C_, 256, 0, stream>>>(outp, x, gamma_p, bnp_w, bnp_b, yf, ybf);
    // 4) CAM energy y y^T
    cam_energy_kernel<<<B_ * 16, 32, 0, stream>>>(ybf, ec);
    // 5) CAM softmax
    cam_softmax_kernel<<<B_, C_, 0, stream>>>(ec, attnc);
    // 6) out_c = attn_c y
    cam_out_kernel<<<B_ * (N_ / 16), 32, 0, stream>>>(attnc, ybf, outc);
    // 7) out = BN(gamma_c * out_c) + y
    bn_add_kernel<<<C_, 256, 0, stream>>>(outc, yf, gamma_c, bnc_w, bnc_b,
                                          (float*)d_out, ybf2);
}
